// MultiHeadSelfAttention_7782480740452
// MI455X (gfx1250) — matrix operand order repaired and checked by reading, not compiled, not run on hardware
//
#include <hip/hip_runtime.h>

typedef _Float16 h16;
typedef __attribute__((ext_vector_type(16))) _Float16 v16h;
typedef __attribute__((ext_vector_type(8)))  _Float16 v8h;
typedef __attribute__((ext_vector_type(8)))  float    v8f;
typedef int v4i __attribute__((vector_size(16)));   // b128 payload type

#define BATCH 2
#define SEQ   2048
#define EMB   1024
#define HEADS 16
#define HDIM  64

// ---------------------------------------------------------------------------
// CDNA5 async global->LDS copy (GLOBAL_LOAD_ASYNC_TO_LDS_B128, ASYNCcnt).
// Builtin signature (from hipcc diagnostic): arg0 = addrspace(1) v4i* (global
// src), arg1 = addrspace(3) v4i* (LDS dst), then imm offset, imm cpol.
// Guarded: falls back to a synchronous v8h copy if the builtin is absent.
// ---------------------------------------------------------------------------
#if defined(__HIP_DEVICE_COMPILE__) && defined(__gfx1250__) && \
    __has_builtin(__builtin_amdgcn_global_load_async_to_lds_b128)
#define HAS_ASYNC_LDS 1
#else
#define HAS_ASYNC_LDS 0
#endif

__device__ __forceinline__ void copy16B_async(const h16* g, h16* l) {
#if HAS_ASYNC_LDS
  __builtin_amdgcn_global_load_async_to_lds_b128(
      (__attribute__((address_space(1))) v4i*)g,
      (__attribute__((address_space(3))) v4i*)l, 0, 0);
#else
  *(v8h*)l = *(const v8h*)g;
#endif
}

__device__ __forceinline__ void async_wait0() {
#if HAS_ASYNC_LDS
#if __has_builtin(__builtin_amdgcn_s_wait_asynccnt)
  __builtin_amdgcn_s_wait_asynccnt(0);
#else
  asm volatile("s_wait_asynccnt 0x0" ::: "memory");
#endif
#endif
}

// ---------------------------------------------------------------------------
// WMMA helpers (CDNA5 f16 16x16x32, fp32 accumulate)
// ---------------------------------------------------------------------------
__device__ __forceinline__ v8f wmma_f16(v16h a, v16h b, v8f c) {
  // (neg_a, A, neg_b, B, c_mod, C, reuse_a, reuse_b)
  return __builtin_amdgcn_wmma_f32_16x16x32_f16(false, a, false, b, (short)0, c,
                                                false, false);
}

// A fragment: row-major LDS tile, 16 rows x 32 K. Lane cl = row, g selects
// K-half. ISA layout: halves 0..7 = K g*8..g*8+7, halves 8..15 = K g*8+16..23.
__device__ __forceinline__ v16h load_a_frag(const h16* base, int ld, int cl, int g) {
  const h16* p = base + cl * ld + g * 8;
  v16h a;
  ((v8h*)&a)[0] = *(const v8h*)(p);
  ((v8h*)&a)[1] = *(const v8h*)(p + 16);
  return a;
}

// B fragment: LDS tile stored [N][K] (column-major B). Lane cl = column N,
// halves 0..7 = K g*8..g*8+7, halves 8..15 = K g*8+16..23 (matches A).
__device__ __forceinline__ v16h load_b_frag(const h16* base, int ld, int cl, int g) {
  const h16* p = base + cl * ld + g * 8;
  v16h b;
  ((v8h*)&b)[0] = *(const v8h*)(p);
  ((v8h*)&b)[1] = *(const v8h*)(p + 16);
  return b;
}

// ---------------------------------------------------------------------------
// Kernel 1: QKV projection  X[4096,1024] @ Wqkv[1024,3072] + bqkv
// Each 64x64 output tile lands entirely in one of Q/K/V for one head.
// Q gets pre-scaled by 1/sqrt(D)=0.125. Output f16 [B,H,S,D].
// (fp32 inputs must convert through VGPRs, so no async staging here.)
// ---------------------------------------------------------------------------
__global__ __launch_bounds__(256) void qkv_gemm(
    const float* __restrict__ X, const float* __restrict__ W,
    const float* __restrict__ bias,
    h16* __restrict__ Qh, h16* __restrict__ Kh, h16* __restrict__ Vh) {
  __shared__ h16 As[64 * 40];   // [M=64][K=32] pitch 40 (80B, 16B aligned)
  __shared__ h16 Bt[64 * 40];   // [N=64][K=32]
  const int tid = threadIdx.x;
  const int lane = tid & 31, cl = lane & 15, g = lane >> 4;
  const int wave = tid >> 5, wm = wave & 3, wn = wave >> 2;
  const int n0 = blockIdx.x * 64;
  const int m0 = blockIdx.y * 64;
  const int ar = tid >> 2, ac = (tid & 3) * 8;   // A staging: 64 rows x 4 thr
  const int bk = tid >> 3, bn = (tid & 7) * 8;   // B staging: 32 rows x 8 thr
  v8f acc0 = {}, acc1 = {};

  for (int k0 = 0; k0 < EMB; k0 += 32) {
    __syncthreads();
    const float* xp = X + (size_t)(m0 + ar) * EMB + k0 + ac;
    __builtin_prefetch(xp + 32, 0, 1);           // global_prefetch_b8
    v8h av;
    #pragma unroll
    for (int i = 0; i < 8; ++i) av[i] = (h16)xp[i];
    *(v8h*)(As + ar * 40 + ac) = av;
    const float* wp = W + (size_t)(k0 + bk) * (3 * EMB) + n0 + bn;
    __builtin_prefetch(wp + (size_t)32 * 3 * EMB, 0, 1);
    #pragma unroll
    for (int i = 0; i < 8; ++i) Bt[(bn + i) * 40 + bk] = (h16)wp[i];
    __syncthreads();
    v16h a  = load_a_frag(As + (wm * 16) * 40, 40, cl, g);
    v16h b0 = load_b_frag(Bt + (wn * 32) * 40, 40, cl, g);
    v16h b1 = load_b_frag(Bt + (wn * 32 + 16) * 40, 40, cl, g);
    acc0 = wmma_f16(a, b0, acc0);
    acc1 = wmma_f16(a, b1, acc1);
  }

  const int part = (n0 % 192) >> 6;       // 0=Q 1=K 2=V (uniform per block)
  const int head = n0 / 192;
  h16* dst = (part == 0) ? Qh : (part == 1) ? Kh : Vh;
  const float scale = (part == 0) ? 0.125f : 1.0f;
  #pragma unroll
  for (int sub = 0; sub < 2; ++sub) {
    const v8f& c = sub ? acc1 : acc0;
    const int n = n0 + wn * 32 + sub * 16 + cl;
    const int d = n & 63;
    const float bv = bias[n];
    #pragma unroll
    for (int r = 0; r < 8; ++r) {
      const int row = m0 + wm * 16 + r + 8 * g;   // flat B*S row
      const int bb = row >> 11, s = row & (SEQ - 1);
      dst[((size_t)(bb * HEADS + head) * SEQ + s) * HDIM + d] =
          (h16)((c[r] + bv) * scale);
    }
  }
}

// ---------------------------------------------------------------------------
// Kernel 2: attention for one (b,h,qtile64). Phase 1: WMMA scores + online
// row max/sum (ALiBi reduces to +slope*j since -slope*i cancels in softmax).
// Phase 2: recompute scores, emit normalized attn (written once), accumulate
// O = P @ V with WMMA. Q/K tiles are staged with async global->LDS DMA.
// ---------------------------------------------------------------------------
__global__ __launch_bounds__(256) void attention(
    const h16* __restrict__ Qh, const h16* __restrict__ Kh,
    const h16* __restrict__ Vh, h16* __restrict__ Oh,
    float* __restrict__ attn) {
  __shared__ h16 Qs[64 * 80];
  __shared__ h16 Ks[64 * 80];
  __shared__ h16 Vt[64 * 80];            // transposed: [d][k]
  __shared__ h16 Ps[64 * 80];
  __shared__ float mrow[64], lrow[64];
  __shared__ float pmax[64 * 2], psum[64 * 2];
  const int tid = threadIdx.x;
  const int lane = tid & 31, cl = lane & 15, g = lane >> 4;
  const int wave = tid >> 5, wm = wave & 3, wn = wave >> 2;
  const int qt = blockIdx.x, head = blockIdx.y, bb = blockIdx.z;
  const float slope = exp2f(-0.5f * (float)(head + 1));   // H=16 ALiBi slopes
  const size_t bh = (size_t)(bb * HEADS + head);
  const h16* Qg = Qh + (bh * SEQ + (size_t)qt * 64) * HDIM;
  const h16* Kg = Kh + bh * SEQ * HDIM;
  const h16* Vg = Vh + bh * SEQ * HDIM;
  const int sr = tid >> 2, sc = (tid & 3) * 16;   // 64 rows x 4 thr x 16 h

  // stage Q tile once (async DMA; completion covered by first loop's wait)
  copy16B_async(Qg + sr * HDIM + sc,     Qs + sr * 80 + sc);
  copy16B_async(Qg + sr * HDIM + sc + 8, Qs + sr * 80 + sc + 8);
  if (tid < 64) { mrow[tid] = -3.0e38f; lrow[tid] = 0.0f; }

  // ---------------- phase 1: stats ----------------
  for (int kt = 0; kt < SEQ / 64; ++kt) {
    __syncthreads();
    const h16* kg = Kg + (size_t)(kt * 64 + sr) * HDIM + sc;
    copy16B_async(kg,     Ks + sr * 80 + sc);
    copy16B_async(kg + 8, Ks + sr * 80 + sc + 8);
    async_wait0();
    __syncthreads();
    v8f s0 = {}, s1 = {};
    #pragma unroll
    for (int dstep = 0; dstep < HDIM; dstep += 32) {
      v16h a  = load_a_frag(Qs + (wm * 16) * 80 + dstep, 80, cl, g);
      v16h b0 = load_b_frag(Ks + (wn * 32) * 80 + dstep, 80, cl, g);
      v16h b1 = load_b_frag(Ks + (wn * 32 + 16) * 80 + dstep, 80, cl, g);
      s0 = wmma_f16(a, b0, s0);
      s1 = wmma_f16(a, b1, s1);
    }
    const float bias0 = slope * (float)(kt * 64 + wn * 32 + cl);
    const float bias1 = bias0 + slope * 16.0f;
    float rmax[8];
    #pragma unroll
    for (int r = 0; r < 8; ++r) {
      s0[r] += bias0; s1[r] += bias1;
      rmax[r] = fmaxf(s0[r], s1[r]);
    }
    #pragma unroll
    for (int off = 1; off < 16; off <<= 1) {
      #pragma unroll
      for (int r = 0; r < 8; ++r)
        rmax[r] = fmaxf(rmax[r], __shfl_xor(rmax[r], off, 32));
    }
    if (cl == 0) {
      #pragma unroll
      for (int r = 0; r < 8; ++r)
        pmax[(wm * 16 + 8 * g + r) * 2 + wn] = rmax[r];
    }
    __syncthreads();
    if (tid < 64) {
      const float tm = fmaxf(pmax[tid * 2], pmax[tid * 2 + 1]);
      const float mo = mrow[tid];
      const float mn = fmaxf(mo, tm);
      mrow[tid] = mn;
      lrow[tid] *= __expf(mo - mn);
    }
    __syncthreads();
    float rsum[8];
    #pragma unroll
    for (int r = 0; r < 8; ++r) {
      const float mn = mrow[wm * 16 + 8 * g + r];
      rsum[r] = __expf(s0[r] - mn) + __expf(s1[r] - mn);
    }
    #pragma unroll
    for (int off = 1; off < 16; off <<= 1) {
      #pragma unroll
      for (int r = 0; r < 8; ++r)
        rsum[r] += __shfl_xor(rsum[r], off, 32);
    }
    if (cl == 0) {
      #pragma unroll
      for (int r = 0; r < 8; ++r)
        psum[(wm * 16 + 8 * g + r) * 2 + wn] = rsum[r];
    }
    __syncthreads();
    if (tid < 64) lrow[tid] += psum[tid * 2] + psum[tid * 2 + 1];
  }
  __syncthreads();

  // ---------------- phase 2: emit attn + O = P@V ----------------
  v8f o0 = {}, o1 = {};
  float* attnBase = attn + (bh * SEQ + (size_t)qt * 64) * SEQ;
  for (int kt = 0; kt < SEQ / 64; ++kt) {
    __syncthreads();
    const h16* kg = Kg + (size_t)(kt * 64 + sr) * HDIM + sc;
    copy16B_async(kg,     Ks + sr * 80 + sc);
    copy16B_async(kg + 8, Ks + sr * 80 + sc + 8);
    { // stage V transposed: Vt[d][k]  (transpose must go through VGPRs)
      const h16* vp = Vg + (size_t)(kt * 64 + sr) * HDIM + sc;
      #pragma unroll
      for (int i = 0; i < 16; ++i) Vt[(sc + i) * 80 + sr] = vp[i];
    }
    async_wait0();
    __syncthreads();
    v8f s0 = {}, s1 = {};
    #pragma unroll
    for (int dstep = 0; dstep < HDIM; dstep += 32) {
      v16h a  = load_a_frag(Qs + (wm * 16) * 80 + dstep, 80, cl, g);
      v16h b0 = load_b_frag(Ks + (wn * 32) * 80 + dstep, 80, cl, g);
      v16h b1 = load_b_frag(Ks + (wn * 32 + 16) * 80 + dstep, 80, cl, g);
      s0 = wmma_f16(a, b0, s0);
      s1 = wmma_f16(a, b1, s1);
    }
    const float bias0 = slope * (float)(kt * 64 + wn * 32 + cl);
    const float bias1 = bias0 + slope * 16.0f;
    const int c0 = kt * 64 + wn * 32 + cl, c1 = c0 + 16;
    #pragma unroll
    for (int r = 0; r < 8; ++r) {
      const int rl = wm * 16 + 8 * g + r;
      const float mn  = mrow[rl];
      const float inv = 1.0f / lrow[rl];
      const float p0 = __expf(s0[r] + bias0 - mn) * inv;
      const float p1 = __expf(s1[r] + bias1 - mn) * inv;
      attnBase[(size_t)rl * SEQ + c0] = p0;
      attnBase[(size_t)rl * SEQ + c1] = p1;
      Ps[rl * 80 + (wn * 32 + cl)]      = (h16)p0;
      Ps[rl * 80 + (wn * 32 + 16 + cl)] = (h16)p1;
    }
    __syncthreads();
    #pragma unroll
    for (int kstep = 0; kstep < 64; kstep += 32) {
      v16h a  = load_a_frag(Ps + (wm * 16) * 80 + kstep, 80, cl, g);
      v16h b0 = load_b_frag(Vt + (wn * 32) * 80 + kstep, 80, cl, g);
      v16h b1 = load_b_frag(Vt + (wn * 32 + 16) * 80 + kstep, 80, cl, g);
      o0 = wmma_f16(a, b0, o0);
      o1 = wmma_f16(a, b1, o1);
    }
  }
  // write O (f16) into [B*S][E] at column head*64+d for the final GEMM
  #pragma unroll
  for (int sub = 0; sub < 2; ++sub) {
    const v8f& c = sub ? o1 : o0;
    const int d = wn * 32 + sub * 16 + cl;
    #pragma unroll
    for (int r = 0; r < 8; ++r) {
      const int rl = wm * 16 + 8 * g + r;
      const size_t row = (size_t)bb * SEQ + qt * 64 + rl;
      Oh[row * EMB + head * HDIM + d] = (h16)c[r];
    }
  }
}

// ---------------------------------------------------------------------------
// Kernel 3: output projection  O[4096,1024](f16) @ Wout[1024,1024] + bout
// A-tile (already f16) staged via async global->LDS DMA.
// ---------------------------------------------------------------------------
__global__ __launch_bounds__(256) void out_gemm(
    const h16* __restrict__ Oh, const float* __restrict__ W,
    const float* __restrict__ bias, float* __restrict__ out) {
  __shared__ h16 As[64 * 40];
  __shared__ h16 Bt[64 * 40];
  const int tid = threadIdx.x;
  const int lane = tid & 31, cl = lane & 15, g = lane >> 4;
  const int wave = tid >> 5, wm = wave & 3, wn = wave >> 2;
  const int n0 = blockIdx.x * 64;
  const int m0 = blockIdx.y * 64;
  const int ar = tid >> 2, ac = (tid & 3) * 8;
  const int bk = tid >> 3, bn = (tid & 7) * 8;
  v8f acc0 = {}, acc1 = {};

  for (int k0 = 0; k0 < EMB; k0 += 32) {
    __syncthreads();
    copy16B_async(Oh + (size_t)(m0 + ar) * EMB + k0 + ac, As + ar * 40 + ac);
    const float* wp = W + (size_t)(k0 + bk) * EMB + n0 + bn;
    __builtin_prefetch(wp + (size_t)32 * EMB, 0, 1);
    #pragma unroll
    for (int i = 0; i < 8; ++i) Bt[(bn + i) * 40 + bk] = (h16)wp[i];
    async_wait0();
    __syncthreads();
    v16h a  = load_a_frag(As + (wm * 16) * 40, 40, cl, g);
    v16h b0 = load_b_frag(Bt + (wn * 32) * 40, 40, cl, g);
    v16h b1 = load_b_frag(Bt + (wn * 32 + 16) * 40, 40, cl, g);
    acc0 = wmma_f16(a, b0, acc0);
    acc1 = wmma_f16(a, b1, acc1);
  }
  #pragma unroll
  for (int sub = 0; sub < 2; ++sub) {
    const v8f& c = sub ? acc1 : acc0;
    const int n = n0 + wn * 32 + sub * 16 + cl;
    const float bv = bias[n];
    #pragma unroll
    for (int r = 0; r < 8; ++r) {
      const int row = m0 + wm * 16 + r + 8 * g;
      out[(size_t)row * EMB + n] = c[r] + bv;
    }
  }
}

// ---------------------------------------------------------------------------
extern "C" void kernel_launch(void* const* d_in, const int* in_sizes, int n_in,
                              void* d_out, int out_size, void* d_ws,
                              size_t ws_size, hipStream_t stream) {
  (void)in_sizes; (void)n_in; (void)out_size; (void)ws_size;
  const float* x    = (const float*)d_in[0];
  const float* Wqkv = (const float*)d_in[1];
  const float* bqkv = (const float*)d_in[2];
  const float* Wout = (const float*)d_in[3];
  const float* bout = (const float*)d_in[4];
  float* out  = (float*)d_out;
  float* attn = out + (size_t)BATCH * SEQ * EMB;   // tuple order: (out, attn)

  const size_t perT = (size_t)BATCH * HEADS * SEQ * HDIM;  // 4 Mi elems
  h16* Qh = (h16*)d_ws;
  h16* Kh = Qh + perT;
  h16* Vh = Kh + perT;
  h16* Oh = Vh + perT;   // [B*S][E] f16, total ws = 32 MiB

  dim3 blk(256);
  qkv_gemm<<<dim3(3 * EMB / 64, BATCH * SEQ / 64), blk, 0, stream>>>(
      x, Wqkv, bqkv, Qh, Kh, Vh);
  attention<<<dim3(SEQ / 64, HEADS, BATCH), blk, 0, stream>>>(
      Qh, Kh, Vh, Oh, attn);
  out_gemm<<<dim3(EMB / 64, BATCH * SEQ / 64), blk, 0, stream>>>(
      Oh, Wout, bout, out);
}
